// MultiHeadAttention_31095563223708
// MI455X (gfx1250) — matrix operand order repaired and checked by reading, not compiled, not run on hardware
//
#include <hip/hip_runtime.h>
#include <math.h>

typedef __attribute__((ext_vector_type(16))) _Float16 v16h;
typedef __attribute__((ext_vector_type(8)))  float    v8f;
typedef _Float16 half_t;

#define HIDDEN 1024
#define HEADS  16
#define HEAD   64
#define BATCH  2
#define SEQ    2048
#define MTOT   (BATCH * SEQ)   // 4096 rows

union FragU { uint4 u[2]; v16h h; };

// Load a 16-half WMMA fragment from two 16-byte chunks.
static __device__ inline v16h load_frag(const half_t* p0, const half_t* p1) {
  FragU f;
  f.u[0] = *reinterpret_cast<const uint4*>(p0);
  f.u[1] = *reinterpret_cast<const uint4*>(p1);
  return f.h;
}

static __device__ inline v8f wmma_f16(v16h a, v16h b, v8f c) {
  // D = A(16x32) * B(32x16) + C, f32 accumulate
  return __builtin_amdgcn_wmma_f32_16x16x32_f16(
      /*neg_a=*/false, a, /*neg_b=*/false, b,
      /*c_mod=*/(short)0, c, /*reuse_a=*/false, /*reuse_b=*/false);
}

// Async global -> LDS (16B per lane), tracked by ASYNCcnt.
// Flat pointers to LDS carry the LDS byte offset in their low 32 bits.
static __device__ inline void async_b128(uint32_t lds, const half_t* g) {
  asm volatile("global_load_async_to_lds_b128 %0, %1, off"
               :: "v"(lds), "v"((unsigned long long)(uintptr_t)g)
               : "memory");
}
static __device__ inline void wait_async0() {
  asm volatile("s_wait_asynccnt 0x0" ::: "memory");
}

// XOR-butterfly within each 16-lane row via v_permlane16_b32 (pure VALU,
// no LDS): selector nibbles give src lane for lanes 0-7 (lo) / 8-15 (hi);
// the same pattern applies to lanes 16-31.
template <int XORM>
static __device__ inline float xor16(float v) {
  unsigned lo, hi;
  if (XORM == 1)      { lo = 0x67452301u; hi = 0xEFCDAB89u; }
  else if (XORM == 2) { lo = 0x45670123u; hi = 0xCDEF89ABu; }
  else if (XORM == 4) { lo = 0x01234567u; hi = 0x89ABCDEFu; }
  else                { lo = 0xFEDCBA98u; hi = 0x76543210u; }  // XORM == 8
  unsigned s = __builtin_bit_cast(unsigned, v);
  unsigned r = __builtin_amdgcn_permlane16(s, s, lo, hi, false, false);
  return __builtin_bit_cast(float, r);
}
static __device__ inline float rowmax16(float v) {
  v = fmaxf(v, xor16<1>(v));
  v = fmaxf(v, xor16<2>(v));
  v = fmaxf(v, xor16<4>(v));
  v = fmaxf(v, xor16<8>(v));
  return v;
}
static __device__ inline float rowsum16(float v) {
  v += xor16<1>(v);
  v += xor16<2>(v);
  v += xor16<4>(v);
  v += xor16<8>(v);
  return v;
}

// ---------------------------------------------------------------- convert
__global__ void cvt_f32_to_f16(const float* __restrict__ src,
                               half_t* __restrict__ dst, int n) {
  int i = blockIdx.x * blockDim.x + threadIdx.x;
  int stride = gridDim.x * blockDim.x;
  for (; i < n; i += stride) dst[i] = (half_t)src[i];
}

// ---------------------------------------------------------------- GEMM
// C[M,N] = A16[M,1024] @ B16[N,1024]^T + bias
// MODE 0: scatter to q16 (pre-scaled by 1/sqrt(d)) / k16 ([B,h,S,d] f16)
//         and vt16 ([B,h,d,S] f16)
// MODE 1: store f32 to outF ([M,N])
// Block tile 128x128, 8 waves (4M x 2N), wave tile 32x64, K-step 64,
// double-buffered async LDS staging (ASYNCcnt).
template <int MODE>
__global__ __launch_bounds__(256)
void gemm_wmma(const half_t* __restrict__ A, const half_t* __restrict__ Bm,
               const float* __restrict__ bias, int N,
               float* __restrict__ outF, half_t* __restrict__ q16,
               half_t* __restrict__ k16, half_t* __restrict__ vt16) {
  const int K = 1024;
  const int PITCH = 72;  // 64 halves + 8 pad (144B rows, 16B aligned)
  __shared__ __align__(16) half_t Asm[2][128 * PITCH];
  __shared__ __align__(16) half_t Bsm[2][128 * PITCH];

  const int tid = threadIdx.x;
  const int lane = tid & 31, wid = tid >> 5;
  const int wave_m = wid >> 1, wave_n = wid & 1;   // 4 x 2 waves
  const int gm0 = blockIdx.y * 128;
  const int gn0 = blockIdx.x * 128;

  v8f acc[2][4] = {};

  // staging: 256 threads, each moves 64B of A and 64B of B per stage
  const int row2 = tid >> 1;            // 0..127
  const int col2 = (tid & 1) * 32;      // half offset within 64

  auto stage = [&](int k0, int buf) {
    const half_t* ga = A + (size_t)(gm0 + row2) * K + k0 + col2;
    const half_t* gb = Bm + (size_t)(gn0 + row2) * K + k0 + col2;
    const uint32_t la = (uint32_t)(uintptr_t)&Asm[buf][row2 * PITCH + col2];
    const uint32_t lb = (uint32_t)(uintptr_t)&Bsm[buf][row2 * PITCH + col2];
#pragma unroll
    for (int c = 0; c < 4; ++c) {
      async_b128(la + c * 16, ga + c * 8);
      async_b128(lb + c * 16, gb + c * 8);
    }
  };

  int buf = 0;
  stage(0, 0);
  for (int k0 = 0; k0 < K; k0 += 64) {
    wait_async0();        // my async stores for `buf` have landed in LDS
    __syncthreads();      // everyone's landed; prior reads of buf^1 done
    if (k0 + 64 < K) stage(k0 + 64, buf ^ 1);

    const half_t* As = &Asm[buf][0];
    const half_t* Bs = &Bsm[buf][0];
#pragma unroll
    for (int t = 0; t < 2; ++t) {   // two K=32 chunks
      v16h af[2], bf[4];
#pragma unroll
      for (int i = 0; i < 2; ++i) {
        // A frag: lane m holds K {c..c+7, 16+c..23+c}, c = (lane>=16)*8
        const half_t* base =
            As + (wave_m * 32 + i * 16 + (lane & 15)) * PITCH + t * 32;
        const int c = (lane >> 4) * 8;
        af[i] = load_frag(base + c, base + 16 + c);
      }
#pragma unroll
      for (int j = 0; j < 4; ++j) {
        // B frag: lane n holds K {c..c+7, 16+c..23+c}, c = (lane>=16)*8
        const half_t* base =
            Bs + (wave_n * 64 + j * 16 + (lane & 15)) * PITCH + t * 32;
        const int c = (lane >> 4) * 8;
        bf[j] = load_frag(base + c, base + 16 + c);
      }
#pragma unroll
      for (int i = 0; i < 2; ++i)
#pragma unroll
        for (int j = 0; j < 4; ++j)
          acc[i][j] = wmma_f16(af[i], bf[j], acc[i][j]);
    }
    buf ^= 1;
  }

#pragma unroll
  for (int i = 0; i < 2; ++i) {
#pragma unroll
    for (int j = 0; j < 4; ++j) {
#pragma unroll
      for (int r = 0; r < 8; ++r) {
        const int m = gm0 + wave_m * 32 + i * 16 + (lane >> 4) * 8 + r;
        const int n = gn0 + wave_n * 64 + j * 16 + (lane & 15);
        const float v = acc[i][j][r] + bias[n];
        if (MODE == 1) {
          outF[(size_t)m * N + n] = v;
        } else {
          const int which = n >> 10, h = (n >> 6) & 15, dd = n & 63;
          const int b = m >> 11, s = m & (SEQ - 1);
          const int bh = b * HEADS + h;
          if (which == 0)  // fold attention scale 1/sqrt(64) into Q
            q16[((size_t)bh * SEQ + s) * HEAD + dd] = (half_t)(v * 0.125f);
          else if (which == 1)
            k16[((size_t)bh * SEQ + s) * HEAD + dd] = (half_t)v;
          else
            vt16[((size_t)bh * HEAD + dd) * SEQ + s] = (half_t)v;
        }
      }
    }
  }
}

// ---------------------------------------------------------------- attention
// One wave per 16-query tile, flash-style online softmax over 32-key chunks.
// K/V fragments are software-pipelined one chunk ahead so global latency is
// hidden behind the softmax VALU chain. Row reductions use v_permlane16_b32
// (no LDS traffic). The P-tile LDS round-trip is wave-private: LDS ops are
// in-order within a wave, so a zero-cost wavefront fence replaces barriers.
__global__ __launch_bounds__(256)
void flash_attn(const half_t* __restrict__ q16, const half_t* __restrict__ k16,
                const half_t* __restrict__ vt16, const int* __restrict__ mask,
                half_t* __restrict__ ctx16) {
  __shared__ __align__(16) half_t Plds[8][16 * 40];  // per-wave 16x32, padded

  const int lane = threadIdx.x & 31, wid = threadIdx.x >> 5;
  const int widx = blockIdx.x * 8 + wid;
  const int bh = widx >> 7;             // / (SEQ/16)
  const int q0 = (widx & 127) * 16;
  const int b = bh >> 4, h = bh & 15;

  // resident Q fragments (16 x 64, split K=0..31 / 32..63), pre-scaled
  v16h aq[2];
#pragma unroll
  for (int t = 0; t < 2; ++t) {
    const half_t* base =
        q16 + ((size_t)bh * SEQ + q0 + (lane & 15)) * HEAD + t * 32;
    const int c = (lane >> 4) * 8;
    aq[t] = load_frag(base + c, base + 16 + c);
  }

  auto load_k = [&](int kk, v16h bk[2][2]) {
#pragma unroll
    for (int j = 0; j < 2; ++j)
#pragma unroll
      for (int t = 0; t < 2; ++t) {
        const half_t* base =
            k16 + ((size_t)bh * SEQ + kk + j * 16 + (lane & 15)) * HEAD +
            t * 32 + (lane >> 4) * 8;
        bk[j][t] = load_frag(base, base + 16);
      }
  };
  auto load_v = [&](int kk, v16h bv[4]) {
#pragma unroll
    for (int n = 0; n < 4; ++n) {
      const half_t* vb = vt16 +
          ((size_t)bh * HEAD + n * 16 + (lane & 15)) * SEQ + kk +
          (lane >> 4) * 8;
      bv[n] = load_frag(vb, vb + 16);
    }
  };

  float m_i[8], l_i[8];
  v8f o[4] = {};
#pragma unroll
  for (int r = 0; r < 8; ++r) { m_i[r] = -3.0e38f; l_i[r] = 0.0f; }

  half_t* P = &Plds[wid][0];

  v16h bk[2][2], bv[4];
  load_k(0, bk);
  load_v(0, bv);

  for (int kk = 0; kk < SEQ; kk += 32) {
    // issue next chunk's fragment loads first; consumed next iteration
    v16h nbk[2][2], nbv[4];
    const bool more = (kk + 32 < SEQ);
    if (more) {
      load_k(kk + 32, nbk);
      load_v(kk + 32, nbv);
    }

    // scores: 16 queries x 32 keys (2 N-tiles x 2 K-chunks)
    v8f sc[2] = {};
#pragma unroll
    for (int j = 0; j < 2; ++j)
#pragma unroll
      for (int t = 0; t < 2; ++t) sc[j] = wmma_f16(aq[t], bk[j][t], sc[j]);

    float mb[2];
#pragma unroll
    for (int j = 0; j < 2; ++j)
      mb[j] = (mask[b * SEQ + kk + j * 16 + (lane & 15)] == 0) ? -1e9f : 0.0f;

#pragma unroll
    for (int r = 0; r < 8; ++r) {
      float s0 = sc[0][r] + mb[0];   // scale pre-folded into Q
      float s1 = sc[1][r] + mb[1];
      const float rm = rowmax16(fmaxf(s0, s1));
      const float mn = fmaxf(m_i[r], rm);
      const float p0 = __expf(s0 - mn);
      const float p1 = __expf(s1 - mn);
      const float ps = rowsum16(p0 + p1);
      const float alpha = __expf(m_i[r] - mn);
      l_i[r] = l_i[r] * alpha + ps;
      m_i[r] = mn;
#pragma unroll
      for (int n = 0; n < 4; ++n) o[n][r] *= alpha;
      // stash P (C-layout) into LDS so it can be re-read in A-fragment layout
      const int row = (lane >> 4) * 8 + r;
      P[row * 40 + (lane & 15)] = (half_t)p0;
      P[row * 40 + 16 + (lane & 15)] = (half_t)p1;
    }
    __builtin_amdgcn_fence(__ATOMIC_ACQ_REL, "wavefront");  // order P st->ld

    // P as A-fragment (16x32), V^T as B-fragments, accumulate O
    {
      const half_t* baseP = P + (lane & 15) * 40;
      const int c = (lane >> 4) * 8;
      v16h ap = load_frag(baseP + c, baseP + 16 + c);
#pragma unroll
      for (int n = 0; n < 4; ++n) o[n] = wmma_f16(ap, bv[n], o[n]);
    }
    __builtin_amdgcn_fence(__ATOMIC_ACQ_REL, "wavefront");  // order ld->next st

    if (more) {
#pragma unroll
      for (int j = 0; j < 2; ++j)
#pragma unroll
        for (int t = 0; t < 2; ++t) bk[j][t] = nbk[j][t];
#pragma unroll
      for (int n = 0; n < 4; ++n) bv[n] = nbv[n];
    }
  }

  // normalize and store ctx as [B,S,HIDDEN] f16 for the output projection
#pragma unroll
  for (int r = 0; r < 8; ++r) {
    const float inv = 1.0f / l_i[r];
    const int srow = q0 + (lane >> 4) * 8 + r;
#pragma unroll
    for (int n = 0; n < 4; ++n) {
      ctx16[((size_t)b * SEQ + srow) * HIDDEN + h * HEAD + n * 16 +
            (lane & 15)] = (half_t)(o[n][r] * inv);
    }
  }
}

// ---------------------------------------------------------------- launch
extern "C" void kernel_launch(void* const* d_in, const int* in_sizes, int n_in,
                              void* d_out, int out_size, void* d_ws,
                              size_t ws_size, hipStream_t stream) {
  (void)in_sizes; (void)n_in; (void)out_size; (void)ws_size;
  const float* x     = (const float*)d_in[0];
  const int*   mask  = (const int*)d_in[1];
  const float* qkv_w = (const float*)d_in[2];
  const float* qkv_b = (const float*)d_in[3];
  const float* out_w = (const float*)d_in[4];
  const float* out_b = (const float*)d_in[5];
  float* out = (float*)d_out;

  char* ws = (char*)d_ws;
  size_t off = 0;
  auto alloc = [&](size_t bytes) -> char* {
    char* p = ws + off;
    off += (bytes + 255) & ~(size_t)255;
    return p;
  };
  half_t* x16   = (half_t*)alloc((size_t)MTOT * HIDDEN * 2);
  half_t* qw16  = (half_t*)alloc((size_t)3 * HIDDEN * HIDDEN * 2);
  half_t* ow16  = (half_t*)alloc((size_t)HIDDEN * HIDDEN * 2);
  half_t* q16   = (half_t*)alloc((size_t)MTOT * HIDDEN * 2);
  half_t* k16   = (half_t*)alloc((size_t)MTOT * HIDDEN * 2);
  half_t* vt16  = (half_t*)alloc((size_t)MTOT * HIDDEN * 2);
  half_t* ctx16 = (half_t*)alloc((size_t)MTOT * HIDDEN * 2);

  cvt_f32_to_f16<<<2048, 256, 0, stream>>>(x, x16, MTOT * HIDDEN);
  cvt_f32_to_f16<<<2048, 256, 0, stream>>>(qkv_w, qw16, 3 * HIDDEN * HIDDEN);
  cvt_f32_to_f16<<<1024, 256, 0, stream>>>(out_w, ow16, HIDDEN * HIDDEN);

  // QKV projection: M=4096, N=3072
  gemm_wmma<0><<<dim3(3 * HIDDEN / 128, MTOT / 128), 256, 0, stream>>>(
      x16, qw16, qkv_b, 3 * HIDDEN, nullptr, q16, k16, vt16);

  // attention: B*h*(S/16) = 4096 wave tiles, 8 waves/block
  flash_attn<<<(BATCH * HEADS * (SEQ / 16)) / 8, 256, 0, stream>>>(
      q16, k16, vt16, mask, ctx16);

  // output projection: M=4096, N=1024, f32 out + bias
  gemm_wmma<1><<<dim3(HIDDEN / 128, MTOT / 128), 256, 0, stream>>>(
      ctx16, ow16, out_b, HIDDEN, out, nullptr, nullptr, nullptr);
}
